// Janus4Block_56066503082207
// MI455X (gfx1250) — matrix operand order repaired and checked by reading, not compiled, not run on hardware
//
#include <hip/hip_runtime.h>
#include <cstdint>
#include <cstddef>

// ---------------------------------------------------------------------------
// Types for CDNA5 WMMA (gfx1250, wave32)
// ---------------------------------------------------------------------------
typedef __bf16 bf16_t;
typedef __attribute__((ext_vector_type(16))) __bf16 v16bf;
typedef __attribute__((ext_vector_type(8)))  float  v8f;
typedef __attribute__((ext_vector_type(8)))  unsigned int v8u;

#define NEGV   (-1.0e9f)
#define RMSEPS 1.1920928955078125e-07f   // finfo(float32).eps

__device__ __forceinline__ v8f wmma_bf16(v16bf a, v16bf b, v8f c) {
    // D = A(16x32 bf16) * B(32x16 bf16) + C(16x16 f32)
    return __builtin_amdgcn_wmma_f32_16x16x32_bf16(false, a, false, b, (short)0, c, false, false);
}

__device__ __forceinline__ v16bf ld_frag_2x8(const bf16_t* p0, const bf16_t* p1) {
    uint4 lo = *(const uint4*)p0;
    uint4 hi = *(const uint4*)p1;
    v8u u = {lo.x, lo.y, lo.z, lo.w, hi.x, hi.y, hi.z, hi.w};
    return __builtin_bit_cast(v16bf, u);
}

// A fragment (16x32, 16-bit): lane m=lane&15; lanes<16: K=e (e<8), K=e+8 (e>=8 -> +16 base)
// ISA layout: lanes 0-15 -> K 0..7 & 16..23 ; lanes 16-31 -> K 8..15 & 24..31
__device__ __forceinline__ v16bf frag_a(const bf16_t* tile, int ld, int lane) {
    int m  = lane & 15;
    int ko = (lane >> 4) << 3;       // 0 or 8
    const bf16_t* p = tile + m * ld + ko;
    return ld_frag_2x8(p, p + 16);
}

// B fragment (32x16, 16-bit) read from N-major storage Bt[n][k] (stride ld elems).
// lane n=lane&15; lanes 0-15 hold K 0..7 & 16..23, lanes 16-31 hold K 8..15 & 24..31 (same as A)
__device__ __forceinline__ v16bf frag_b(const bf16_t* tileT, int ld, int lane) {
    int n  = lane & 15;
    int ko = (lane >> 4) << 3;       // 0 or 8
    const bf16_t* p = tileT + n * ld + ko;
    return ld_frag_2x8(p, p + 16);
}

__device__ __forceinline__ v8f v8f_zero() {
    v8f z = {0.f, 0.f, 0.f, 0.f, 0.f, 0.f, 0.f, 0.f};
    return z;
}

// ---------------------------------------------------------------------------
// fp32 -> bf16 convert (weights / activations)
// ---------------------------------------------------------------------------
__global__ __launch_bounds__(256) void cvt_f32_bf16_kernel(
        const float* __restrict__ in, bf16_t* __restrict__ out, int n) {
    int i = blockIdx.x * blockDim.x + threadIdx.x;
    int stride = gridDim.x * blockDim.x;
    for (; i < n; i += stride) out[i] = (bf16_t)in[i];
}

// ---------------------------------------------------------------------------
// RMSNorm: one block per row. Emits bf16 xn and (optionally) 1/rms per row.
// ---------------------------------------------------------------------------
__global__ __launch_bounds__(256) void rmsnorm_kernel(
        const float* __restrict__ x, const float* __restrict__ w,
        bf16_t* __restrict__ xn, float* __restrict__ invr, int Cc) {
    const int row = blockIdx.x;
    const float* xr = x + (size_t)row * Cc;
    float ss = 0.f;
    for (int i = threadIdx.x; i < Cc; i += blockDim.x) { float v = xr[i]; ss += v * v; }
    ss += __shfl_xor(ss, 1);  ss += __shfl_xor(ss, 2);
    ss += __shfl_xor(ss, 4);  ss += __shfl_xor(ss, 8);
    ss += __shfl_xor(ss, 16);
    __shared__ float red[8];
    if ((threadIdx.x & 31) == 0) red[threadIdx.x >> 5] = ss;
    __syncthreads();
    float tot = 0.f;
#pragma unroll
    for (int i = 0; i < 8; ++i) tot += red[i];
    float inv = rsqrtf(tot / (float)Cc + RMSEPS);
    if (invr != nullptr && threadIdx.x == 0) invr[row] = inv;
    for (int i = threadIdx.x; i < Cc; i += blockDim.x)
        xn[(size_t)row * Cc + i] = (bf16_t)(xr[i] * inv * w[i]);
}

// diff[b,t,:] = xn[b,t,:] - xn[b,t-1,:]  (computed in f32 from x and 1/rms)
__global__ __launch_bounds__(256) void diff_kernel(
        const float* __restrict__ x, const float* __restrict__ w,
        const float* __restrict__ invr, bf16_t* __restrict__ dif, int T_, int Cc) {
    const int row = blockIdx.x;
    const int t = row % T_;
    const float* xr = x + (size_t)row * Cc;
    const float i0 = invr[row];
    const float i1 = (t > 0) ? invr[row - 1] : 0.f;
    for (int i = threadIdx.x; i < Cc; i += blockDim.x) {
        float cur = xr[i] * i0 * w[i];
        float prv = (t > 0) ? xr[i - Cc] * i1 * w[i] : 0.f;
        dif[(size_t)row * Cc + i] = (bf16_t)(cur - prv);
    }
}

// energy[row] = mean(|delta_f[row,:]|)
__global__ __launch_bounds__(256) void energy_kernel(
        const float* __restrict__ df, float* __restrict__ e, int Cc) {
    const int row = blockIdx.x;
    const float* r = df + (size_t)row * Cc;
    float s = 0.f;
    for (int i = threadIdx.x; i < Cc; i += blockDim.x) s += fabsf(r[i]);
    s += __shfl_xor(s, 1);  s += __shfl_xor(s, 2);
    s += __shfl_xor(s, 4);  s += __shfl_xor(s, 8);
    s += __shfl_xor(s, 16);
    __shared__ float red[8];
    if ((threadIdx.x & 31) == 0) red[threadIdx.x >> 5] = s;
    __syncthreads();
    if (threadIdx.x == 0) {
        float tot = 0.f;
#pragma unroll
        for (int i = 0; i < 8; ++i) tot += red[i];
        e[row] = tot / (float)Cc;
    }
}

// ---------------------------------------------------------------------------
// Tiled bf16 WMMA GEMM: C[MxN] = A[MxK] * B[KxN], f32 accumulate.
// 256 threads = 8 waves in a 4x2 grid; block tile 128x128, BK=64.
// Epilogue modes:
//   0: bf16 store               1: bf16 + f32 store
//   2: f32 = res + acc          3: bf16 = silu(acc + bias)
//   4: f32 = res + acc + bias
// ---------------------------------------------------------------------------
template <int MODE>
__global__ __launch_bounds__(256) void gemm_bf16_kernel(
        const bf16_t* __restrict__ A, const bf16_t* __restrict__ Bm,
        int M, int N, int K,
        const float* __restrict__ bias, const float* __restrict__ res,
        float* __restrict__ outF, bf16_t* __restrict__ outB) {
    constexpr int BM = 128, BN = 128, BK = 64;
    __shared__ bf16_t As[BM * BK];     // [m][k], stride BK
    __shared__ bf16_t Bs[BN * BK];     // [n][k], stride BK (transposed)

    const int tiles_n = N / BN;
    const int m0 = (blockIdx.x / tiles_n) * BM;
    const int n0 = (blockIdx.x % tiles_n) * BN;
    const int tid = threadIdx.x;
    const int lane = tid & 31;
    const int wid = tid >> 5;
    const int wr = wid & 3;            // wave row (0..3)  -> 32 rows each
    const int wc = wid >> 2;           // wave col (0..1)  -> 64 cols each

    v8f acc[2][4];
#pragma unroll
    for (int i = 0; i < 2; ++i)
#pragma unroll
        for (int j = 0; j < 4; ++j) acc[i][j] = v8f_zero();

    for (int k0 = 0; k0 < K; k0 += BK) {
        // Stage A: 128x64 bf16 (uint4 = 8 elems per chunk)
        for (int i = tid; i < BM * BK / 8; i += 256) {
            int row = i >> 3, c8 = i & 7;
            *(uint4*)&As[row * BK + c8 * 8] =
                *(const uint4*)&A[(size_t)(m0 + row) * K + k0 + c8 * 8];
        }
        // Stage B transposed: global [64 x 128] -> Bs[n][k]
        for (int i = tid; i < BK * BN / 8; i += 256) {
            int kk = i >> 4, c8 = i & 15;
            uint4 v = *(const uint4*)&Bm[(size_t)(k0 + kk) * N + n0 + c8 * 8];
            const bf16_t* e = (const bf16_t*)&v;
#pragma unroll
            for (int j = 0; j < 8; ++j) Bs[(c8 * 8 + j) * BK + kk] = e[j];
        }
        __syncthreads();
#pragma unroll
        for (int s = 0; s < 2; ++s) {
            v16bf a0 = frag_a(As + (wr * 32 + 0)  * BK + s * 32, BK, lane);
            v16bf a1 = frag_a(As + (wr * 32 + 16) * BK + s * 32, BK, lane);
#pragma unroll
            for (int j = 0; j < 4; ++j) {
                v16bf b = frag_b(Bs + (wc * 64 + j * 16) * BK + s * 32, BK, lane);
                acc[0][j] = wmma_bf16(a0, b, acc[0][j]);
                acc[1][j] = wmma_bf16(a1, b, acc[1][j]);
            }
        }
        __syncthreads();
    }

    const int half = lane >> 4;
    const int nn = lane & 15;
#pragma unroll
    for (int i = 0; i < 2; ++i)
#pragma unroll
        for (int j = 0; j < 4; ++j)
#pragma unroll
            for (int r = 0; r < 8; ++r) {
                int row = m0 + wr * 32 + i * 16 + r + half * 8;
                int col = n0 + wc * 64 + j * 16 + nn;
                size_t idx = (size_t)row * N + col;
                float v = acc[i][j][r];
                if constexpr (MODE == 0) {
                    outB[idx] = (bf16_t)v;
                } else if constexpr (MODE == 1) {
                    outB[idx] = (bf16_t)v;
                    outF[idx] = v;
                } else if constexpr (MODE == 2) {
                    outF[idx] = res[idx] + v;
                } else if constexpr (MODE == 3) {
                    float z = v + bias[col];
                    outB[idx] = (bf16_t)(z / (1.f + __expf(-z)));
                } else { // MODE == 4
                    outF[idx] = res[idx] + v + bias[col];
                }
            }
}

// ---------------------------------------------------------------------------
// Fused 4-branch causal flash attention.
// One workgroup = (b, h, 16-query tile); 4 waves, wave w handles branch w:
//   0: softmax(Q K^T / 8) V        1: softmax(Q RK^T / 8) V
//   2: softmax(Q E^T / 8) E        3: softmax(E[q]E[k]/(1+|q-k|)) Delta
// Online softmax over 32-key chunks; combine with gate softmax via LDS atomics.
// ---------------------------------------------------------------------------
__global__ __launch_bounds__(128) void attn4_kernel(
        const bf16_t* __restrict__ qkv,    // [B*T, 3072]
        const bf16_t* __restrict__ rk,     // [B*T, 1024]
        const bf16_t* __restrict__ echo,   // [B*T, 1024]
        const bf16_t* __restrict__ delta,  // [B*T, 1024]
        const float*  __restrict__ energy, // [B*T]
        const float*  __restrict__ gate,   // [H,4]
        bf16_t* __restrict__ y)            // [B*T, 1024]
{
    constexpr int T = 1024, Cc = 1024;
    __shared__ bf16_t Qs[16 * 64];
    __shared__ bf16_t Ks[4][32 * 64];   // [key][dim]
    __shared__ bf16_t Vt[4][64 * 32];   // [dim][key]
    __shared__ bf16_t Ps[4][16 * 32];   // [q][key]
    __shared__ float  Yc[16 * 64];

    const int bid = blockIdx.x;
    const int qt = bid & 63;
    const int h  = (bid >> 6) & 15;
    const int b  = bid >> 10;
    const int q0 = qt * 16;
    const int bT = b * T;

    const int tid = threadIdx.x;
    const int lane = tid & 31;
    const int wid = tid >> 5;
    const int half = lane >> 4;
    const int nn = lane & 15;

    // Stage Q (head slice from qkv) + zero combine buffer
    {
        int row = tid >> 3, c8 = tid & 7;
        *(uint4*)&Qs[row * 64 + c8 * 8] =
            *(const uint4*)&qkv[(size_t)(bT + q0 + row) * 3072 + h * 64 + c8 * 8];
#pragma unroll
        for (int i = 0; i < 8; ++i) Yc[tid * 8 + i] = 0.f;
    }
    __syncthreads();

    // Per-wave (branch) key/value sources
    const bf16_t* kbase; int kld, kcol;
    const bf16_t* vbase; int vld, vcol;
    if (wid == 0)      { kbase = qkv;  kld = 3072; kcol = 1024 + h * 64;
                         vbase = qkv;  vld = 3072; vcol = 2048 + h * 64; }
    else if (wid == 1) { kbase = rk;   kld = 1024; kcol = h * 64;
                         vbase = qkv;  vld = 3072; vcol = 2048 + h * 64; }
    else if (wid == 2) { kbase = echo; kld = 1024; kcol = h * 64;
                         vbase = echo; vld = 1024; vcol = h * 64; }
    else               { kbase = nullptr; kld = 0; kcol = 0;
                         vbase = delta; vld = 1024; vcol = h * 64; }

    // Gate softmax (per head), pick this branch's weight
    float gv;
    {
        float g0 = gate[h * 4 + 0], g1 = gate[h * 4 + 1];
        float g2 = gate[h * 4 + 2], g3 = gate[h * 4 + 3];
        float mg = fmaxf(fmaxf(g0, g1), fmaxf(g2, g3));
        float e0 = __expf(g0 - mg), e1 = __expf(g1 - mg);
        float e2 = __expf(g2 - mg), e3 = __expf(g3 - mg);
        float s = e0 + e1 + e2 + e3;
        float ga[4] = {e0 / s, e1 / s, e2 / s, e3 / s};
        gv = ga[wid];
    }

    // Q fragments (invariant over chunks)
    v16bf aq0 = frag_a(Qs, 64, lane);
    v16bf aq1 = frag_a(Qs + 32, 64, lane);

    float eqr[8];
    if (wid == 3) {
#pragma unroll
        for (int r = 0; r < 8; ++r) eqr[r] = energy[bT + q0 + r + half * 8];
    }

    v8f acc[4];
#pragma unroll
    for (int j = 0; j < 4; ++j) acc[j] = v8f_zero();
    float mrow[8], lrow[8];
#pragma unroll
    for (int r = 0; r < 8; ++r) { mrow[r] = -1.0e30f; lrow[r] = 0.f; }

    const float scale = 0.125f;  // 1/sqrt(64)
    const int nch = (q0 >> 5) + 1;

    for (int c = 0; c < nch; ++c) {
        const int k0 = c * 32;
        // Stage 32-key K chunk [key][dim] (branches 0-2)
        if (wid != 3) {
            const bf16_t* src = kbase + (size_t)(bT + k0 + lane) * kld + kcol;
#pragma unroll
            for (int c8 = 0; c8 < 8; ++c8)
                *(uint4*)&Ks[wid][lane * 64 + c8 * 8] = *(const uint4*)&src[c8 * 8];
        }
        // Stage V chunk transposed [dim][key]
        {
            const bf16_t* src = vbase + (size_t)(bT + k0 + lane) * vld + vcol;
#pragma unroll
            for (int c8 = 0; c8 < 8; ++c8) {
                uint4 v = *(const uint4*)&src[c8 * 8];
                const bf16_t* e = (const bf16_t*)&v;
#pragma unroll
                for (int j = 0; j < 8; ++j) Vt[wid][(c8 * 8 + j) * 32 + lane] = e[j];
            }
        }
        asm volatile("s_wait_dscnt 0" ::: "memory");

        // Scores for two 16-key subtiles
        v8f s0, s1;
        if (wid != 3) {
            s0 = v8f_zero();
            s0 = wmma_bf16(aq0, frag_b(Ks[wid] + 0,            64, lane), s0);
            s0 = wmma_bf16(aq1, frag_b(Ks[wid] + 32,           64, lane), s0);
            s1 = v8f_zero();
            s1 = wmma_bf16(aq0, frag_b(Ks[wid] + 16 * 64,      64, lane), s1);
            s1 = wmma_bf16(aq1, frag_b(Ks[wid] + 16 * 64 + 32, 64, lane), s1);
#pragma unroll
            for (int r = 0; r < 8; ++r) {
                int qi = q0 + r + half * 8;
                s0[r] = (k0 + nn > qi)      ? NEGV : s0[r] * scale;
                s1[r] = (k0 + 16 + nn > qi) ? NEGV : s1[r] * scale;
            }
        } else {
            float ek0 = energy[bT + k0 + nn];
            float ek1 = energy[bT + k0 + 16 + nn];
#pragma unroll
            for (int r = 0; r < 8; ++r) {
                int qi = q0 + r + half * 8;
                int k_a = k0 + nn, k_b = k0 + 16 + nn;
                s0[r] = (k_a > qi) ? NEGV : eqr[r] * ek0 / (1.f + (float)(qi - k_a));
                s1[r] = (k_b > qi) ? NEGV : eqr[r] * ek1 / (1.f + (float)(qi - k_b));
            }
        }

        // Online softmax update (row reductions across 16-lane groups)
        float corr[8];
#pragma unroll
        for (int r = 0; r < 8; ++r) {
            float mx = fmaxf(s0[r], s1[r]);
            mx = fmaxf(mx, __shfl_xor(mx, 1));
            mx = fmaxf(mx, __shfl_xor(mx, 2));
            mx = fmaxf(mx, __shfl_xor(mx, 4));
            mx = fmaxf(mx, __shfl_xor(mx, 8));
            float mnew = fmaxf(mrow[r], mx);
            corr[r] = __expf(mrow[r] - mnew);
            float p0 = __expf(s0[r] - mnew);
            float p1 = __expf(s1[r] - mnew);
            float rs = p0 + p1;
            rs += __shfl_xor(rs, 1);
            rs += __shfl_xor(rs, 2);
            rs += __shfl_xor(rs, 4);
            rs += __shfl_xor(rs, 8);
            lrow[r] = lrow[r] * corr[r] + rs;
            mrow[r] = mnew;
            // stash probabilities into P tile (bf16)
            Ps[wid][(r + half * 8) * 32 + nn]      = (bf16_t)p0;
            Ps[wid][(r + half * 8) * 32 + 16 + nn] = (bf16_t)p1;
        }
#pragma unroll
        for (int j = 0; j < 4; ++j)
#pragma unroll
            for (int r = 0; r < 8; ++r) acc[j][r] *= corr[r];

        asm volatile("s_wait_dscnt 0" ::: "memory");
        v16bf ap = frag_a(Ps[wid], 32, lane);
#pragma unroll
        for (int j = 0; j < 4; ++j) {
            v16bf bv = frag_b(Vt[wid] + (j * 16) * 32, 32, lane);
            acc[j] = wmma_bf16(ap, bv, acc[j]);
        }
    }

    // Normalize, scale by gate, combine across branches
    float inv[8];
#pragma unroll
    for (int r = 0; r < 8; ++r) inv[r] = gv / lrow[r];
#pragma unroll
    for (int j = 0; j < 4; ++j)
#pragma unroll
        for (int r = 0; r < 8; ++r)
            atomicAdd(&Yc[(r + half * 8) * 64 + j * 16 + nn], acc[j][r] * inv[r]);
    __syncthreads();

    // Store combined head output (bf16, feeds w_out GEMM)
    {
        int row = tid >> 3, c8 = tid & 7;
        union { bf16_t hh[8]; uint4 u; } tmp;
#pragma unroll
        for (int i = 0; i < 8; ++i) tmp.hh[i] = (bf16_t)Yc[row * 64 + c8 * 8 + i];
        *(uint4*)&y[(size_t)(bT + q0 + row) * Cc + h * 64 + c8 * 8] = tmp.u;
    }
}

// ---------------------------------------------------------------------------
// Host-side orchestration
// ---------------------------------------------------------------------------
extern "C" void kernel_launch(void* const* d_in, const int* in_sizes, int n_in,
                              void* d_out, int out_size, void* d_ws, size_t ws_size,
                              hipStream_t stream) {
    (void)in_sizes; (void)n_in; (void)out_size; (void)ws_size;
    constexpr int Bb = 2, T = 1024, C = 1024, H = 16, Mm = 4096;
    constexpr int R = Bb * T;  // 2048 rows

    const float* x      = (const float*)d_in[0];
    const float* rms1_w = (const float*)d_in[1];
    const float* w_qkv  = (const float*)d_in[2];
    const float* w_rr   = (const float*)d_in[3];
    const float* w_echo = (const float*)d_in[4];
    const float* w_delta= (const float*)d_in[5];
    const float* gate   = (const float*)d_in[6];
    const float* w_out  = (const float*)d_in[7];
    const float* rms2_w = (const float*)d_in[8];
    const float* w1     = (const float*)d_in[9];
    const float* b1     = (const float*)d_in[10];
    const float* w2     = (const float*)d_in[11];
    const float* b2     = (const float*)d_in[12];
    float* out = (float*)d_out;

    uintptr_t p = (uintptr_t)d_ws;
    auto alloc = [&](size_t bytes) {
        void* r = (void*)p;
        p += (bytes + 255) & ~(size_t)255;
        return r;
    };
    bf16_t* wqkv_b = (bf16_t*)alloc((size_t)C * 3 * C * 2);
    bf16_t* wrr_b  = (bf16_t*)alloc((size_t)C * C * 2);
    bf16_t* wech_b = (bf16_t*)alloc((size_t)C * C * 2);
    bf16_t* wdel_b = (bf16_t*)alloc((size_t)C * C * 2);
    bf16_t* wout_b = (bf16_t*)alloc((size_t)C * C * 2);
    bf16_t* w1_b   = (bf16_t*)alloc((size_t)C * Mm * 2);
    bf16_t* w2_b   = (bf16_t*)alloc((size_t)Mm * C * 2);
    bf16_t* xn_b   = (bf16_t*)alloc((size_t)R * C * 2);
    float*  invr   = (float*) alloc((size_t)R * 4);
    bf16_t* dif_b  = (bf16_t*)alloc((size_t)R * C * 2);
    bf16_t* qkv_b  = (bf16_t*)alloc((size_t)R * 3 * C * 2);
    bf16_t* rk_b   = (bf16_t*)alloc((size_t)R * C * 2);
    bf16_t* ech_b  = (bf16_t*)alloc((size_t)R * C * 2);
    bf16_t* del_b  = (bf16_t*)alloc((size_t)R * C * 2);
    float*  del_f  = (float*) alloc((size_t)R * C * 4);
    float*  energy = (float*) alloc((size_t)R * 4);
    bf16_t* y_b    = (bf16_t*)alloc((size_t)R * C * 2);
    float*  x2_f   = (float*) alloc((size_t)R * C * 4);
    bf16_t* xn2_b  = (bf16_t*)alloc((size_t)R * C * 2);
    bf16_t* h_b    = (bf16_t*)alloc((size_t)R * Mm * 2);

    // 1) Weight conversions to bf16
    cvt_f32_bf16_kernel<<<2048, 256, 0, stream>>>(w_qkv,  wqkv_b, C * 3 * C);
    cvt_f32_bf16_kernel<<<2048, 256, 0, stream>>>(w_rr,   wrr_b,  C * C);
    cvt_f32_bf16_kernel<<<2048, 256, 0, stream>>>(w_echo, wech_b, C * C);
    cvt_f32_bf16_kernel<<<2048, 256, 0, stream>>>(w_delta,wdel_b, C * C);
    cvt_f32_bf16_kernel<<<2048, 256, 0, stream>>>(w_out,  wout_b, C * C);
    cvt_f32_bf16_kernel<<<4096, 256, 0, stream>>>(w1,     w1_b,   C * Mm);
    cvt_f32_bf16_kernel<<<4096, 256, 0, stream>>>(w2,     w2_b,   Mm * C);

    // 2) RMSNorm-1 (+ 1/rms), shifted difference
    rmsnorm_kernel<<<R, 256, 0, stream>>>(x, rms1_w, xn_b, invr, C);
    diff_kernel<<<R, 256, 0, stream>>>(x, rms1_w, invr, dif_b, T, C);

    // 3) Projection GEMMs
    gemm_bf16_kernel<0><<<(R/128) * (3*C/128), 256, 0, stream>>>(
        xn_b, wqkv_b, R, 3*C, C, nullptr, nullptr, nullptr, qkv_b);
    gemm_bf16_kernel<0><<<(R/128) * (C/128), 256, 0, stream>>>(
        xn_b, wrr_b,  R, C, C, nullptr, nullptr, nullptr, rk_b);
    gemm_bf16_kernel<0><<<(R/128) * (C/128), 256, 0, stream>>>(
        xn_b, wech_b, R, C, C, nullptr, nullptr, nullptr, ech_b);
    gemm_bf16_kernel<1><<<(R/128) * (C/128), 256, 0, stream>>>(
        dif_b, wdel_b, R, C, C, nullptr, nullptr, del_f, del_b);

    // 4) Energy = mean |delta_f|
    energy_kernel<<<R, 256, 0, stream>>>(del_f, energy, C);

    // 5) Fused 4-branch attention -> y (bf16)
    attn4_kernel<<<Bb * H * (T / 16), 128, 0, stream>>>(
        qkv_b, rk_b, ech_b, del_b, energy, gate, y_b);

    // 6) x2 = x + y @ w_out
    gemm_bf16_kernel<2><<<(R/128) * (C/128), 256, 0, stream>>>(
        y_b, wout_b, R, C, C, nullptr, x, x2_f, nullptr);

    // 7) RMSNorm-2, MLP
    rmsnorm_kernel<<<R, 256, 0, stream>>>(x2_f, rms2_w, xn2_b, nullptr, C);
    gemm_bf16_kernel<3><<<(R/128) * (Mm/128), 256, 0, stream>>>(
        xn2_b, w1_b, R, Mm, C, b1, nullptr, nullptr, h_b);
    gemm_bf16_kernel<4><<<(R/128) * (C/128), 256, 0, stream>>>(
        h_b, w2_b, R, C, Mm, b2, x2_f, out, nullptr);
}
